// DiscreteDecisionEngine_2980707303712
// MI455X (gfx1250) — matrix operand order repaired and checked by reading, not compiled, not run on hardware
//
#include <hip/hip_runtime.h>
#include <hip/hip_bf16.h>
#include <stddef.h>

// ---------------------------------------------------------------------------
// VQ codebook argmin + gather for MI455X (gfx1250, wave32, WMMA).
// dots computed as split-bf16 (hi+lo) x (hi+lo) with 3 WMMAs -> ~fp32 accuracy
// at bf16-matrix throughput.
// ---------------------------------------------------------------------------

#define LATENT      512
#define KCODES      8192
#define NTOK        32768
#define TOK_PER_BLK 32
#define APAD        520   // padded row stride (ushorts) to dodge LDS bank conflicts

typedef __attribute__((ext_vector_type(16))) __bf16 v16bf;
typedef __attribute__((ext_vector_type(8)))  float  v8f;

union Frag16 {
    uint4          q[2];
    unsigned short u[16];
    v16bf          v;
};
static_assert(sizeof(Frag16) == 32, "frag size");

__device__ __forceinline__ unsigned short f2bf(float f) {
    // round-to-nearest-even truncation to bf16 (inputs are finite gaussians)
    unsigned u = __float_as_uint(f);
    u += 0x7fffu + ((u >> 16) & 1u);
    return (unsigned short)(u >> 16);
}
__device__ __forceinline__ float bf2f(unsigned short h) {
    return __uint_as_float(((unsigned)h) << 16);
}

// ---------------------------------------------------------------------------
// Pass 1: split codebook into bf16 hi/lo and compute exact fp32 ||c||^2.
// One block per codebook row.
// ---------------------------------------------------------------------------
__global__ __launch_bounds__(256) void vq_split_codebook(
    const float* __restrict__ cb,
    unsigned short* __restrict__ chi,
    unsigned short* __restrict__ clo,
    float* __restrict__ csq)
{
    const int code = blockIdx.x;
    const int tid  = threadIdx.x;
    const int lane = tid & 31;
    const int w    = tid >> 5;

    float ss = 0.0f;
    const size_t base = (size_t)code * LATENT;
    for (int k = tid; k < LATENT; k += 256) {
        float f = cb[base + k];
        unsigned short h = f2bf(f);
        chi[base + k] = h;
        clo[base + k] = f2bf(f - bf2f(h));
        ss += f * f;
    }
    // wave32 reduce
    #pragma unroll
    for (int m = 16; m >= 1; m >>= 1) ss += __shfl_xor(ss, m, 32);

    __shared__ float red[8];
    if (lane == 0) red[w] = ss;
    __syncthreads();
    if (tid == 0) {
        float s = 0.0f;
        #pragma unroll
        for (int i = 0; i < 8; ++i) s += red[i];
        csq[code] = s;
    }
}

// ---------------------------------------------------------------------------
// Pass 2: per-token argmin over all 8192 codes + gather winning rows.
// Block = 256 threads = 8 waves; block owns 32 tokens; waves split code tiles.
// ---------------------------------------------------------------------------
__global__ __launch_bounds__(256) void vq_argmin_gather(
    const float* __restrict__ x,
    const float* __restrict__ cb,
    const unsigned short* __restrict__ cbhi,
    const unsigned short* __restrict__ cblo,
    const float* __restrict__ csq,
    float* __restrict__ out)
{
    extern __shared__ char smem[];
    unsigned short* sAhi = (unsigned short*)smem;                 // 32*520 ushorts
    unsigned short* sAlo = sAhi + TOK_PER_BLK * APAD;             // 32*520 ushorts
    float* sBv = (float*)(sAlo + TOK_PER_BLK * APAD);             // [8][32]
    int*   sBi = (int*)(sBv + 8 * 32);                            // [8][32]
    int*   sFin = sBi + 8 * 32;                                   // [32]

    const int tid  = threadIdx.x;
    const int lane = tid & 31;
    const int w    = tid >> 5;
    const int blk  = blockIdx.x;
    const int tok0 = blk * TOK_PER_BLK;

    // ---- stage + split the 32x512 x-tile into LDS (bf16 hi/lo) ----
    for (int e = tid; e < TOK_PER_BLK * LATENT; e += 256) {
        const int tr = e >> 9;          // token row in tile
        const int k  = e & (LATENT - 1);
        float f = x[(size_t)(tok0 + tr) * LATENT + k];
        unsigned short h = f2bf(f);
        sAhi[tr * APAD + k] = h;
        sAlo[tr * APAD + k] = f2bf(f - bf2f(h));
    }
    __syncthreads();

    const int nl   = lane & 15;   // column-within-tile / row-within-frag
    const int half = lane >> 4;   // K-half selector per ISA 16-bit striping

    float bv0[8], bv1[8];
    int   bi0[8], bi1[8];
    #pragma unroll
    for (int r = 0; r < 8; ++r) { bv0[r] = bv1[r] = 3.4e38f; bi0[r] = bi1[r] = 0; }

    // A-fragment LDS base offsets for this lane (K pattern: 8h..8h+7, 8h+16..8h+23)
    const unsigned short* a0h_base = sAhi + nl * APAD + half * 8;
    const unsigned short* a0l_base = sAlo + nl * APAD + half * 8;
    const unsigned short* a1h_base = sAhi + (nl + 16) * APAD + half * 8;
    const unsigned short* a1l_base = sAlo + (nl + 16) * APAD + half * 8;

    for (int ct = w; ct < KCODES / 16; ct += 8) {
        v8f acc0 = {};
        v8f acc1 = {};
        const int code = ct * 16 + nl;
        // B-fragment global base (K pattern: 8h..8h+7, 8h+16..8h+23), bf16 codebook in L2
        const unsigned short* bh = cbhi + (size_t)code * LATENT + half * 8;
        const unsigned short* bl = cblo + (size_t)code * LATENT + half * 8;

        #pragma unroll 2
        for (int ks = 0; ks < LATENT / 32; ++ks) {
            Frag16 bhf, blf, a0h, a0l, a1h, a1l;
            const uint4* p;
            p = (const uint4*)(bh + ks * 32); bhf.q[0] = p[0]; bhf.q[1] = p[2];
            p = (const uint4*)(bl + ks * 32); blf.q[0] = p[0]; blf.q[1] = p[2];

            const unsigned short* ap;
            ap = a0h_base + ks * 32; a0h.q[0] = *(const uint4*)ap; a0h.q[1] = *(const uint4*)(ap + 16);
            ap = a0l_base + ks * 32; a0l.q[0] = *(const uint4*)ap; a0l.q[1] = *(const uint4*)(ap + 16);
            ap = a1h_base + ks * 32; a1h.q[0] = *(const uint4*)ap; a1h.q[1] = *(const uint4*)(ap + 16);
            ap = a1l_base + ks * 32; a1l.q[0] = *(const uint4*)ap; a1l.q[1] = *(const uint4*)(ap + 16);

            // split-bf16 product: hi*hi + lo*hi + hi*lo  (lo*lo ~ 2^-16 relative, dropped)
            acc0 = __builtin_amdgcn_wmma_f32_16x16x32_bf16(false, a0h.v, false, bhf.v, (short)0, acc0, false, false);
            acc1 = __builtin_amdgcn_wmma_f32_16x16x32_bf16(false, a1h.v, false, bhf.v, (short)0, acc1, false, false);
            acc0 = __builtin_amdgcn_wmma_f32_16x16x32_bf16(false, a0l.v, false, bhf.v, (short)0, acc0, false, false);
            acc1 = __builtin_amdgcn_wmma_f32_16x16x32_bf16(false, a1l.v, false, bhf.v, (short)0, acc1, false, false);
            acc0 = __builtin_amdgcn_wmma_f32_16x16x32_bf16(false, a0h.v, false, blf.v, (short)0, acc0, false, false);
            acc1 = __builtin_amdgcn_wmma_f32_16x16x32_bf16(false, a1h.v, false, blf.v, (short)0, acc1, false, false);
        }

        const float cs = csq[code];
        #pragma unroll
        for (int r = 0; r < 8; ++r) {
            float s0 = fmaf(-2.0f, acc0[r], cs);   // ||c||^2 - 2*dot
            if (s0 < bv0[r]) { bv0[r] = s0; bi0[r] = code; }
            float s1 = fmaf(-2.0f, acc1[r], cs);
            if (s1 < bv1[r]) { bv1[r] = s1; bi1[r] = code; }
        }
    }

    // ---- cross-lane argmin over the 16 candidate columns in each half ----
    #pragma unroll
    for (int r = 0; r < 8; ++r) {
        float v0 = bv0[r]; int i0 = bi0[r];
        float v1 = bv1[r]; int i1 = bi1[r];
        #pragma unroll
        for (int m = 1; m < 16; m <<= 1) {
            float ov = __shfl_xor(v0, m, 32); int oi = __shfl_xor(i0, m, 32);
            if (ov < v0 || (ov == v0 && oi < i0)) { v0 = ov; i0 = oi; }
            ov = __shfl_xor(v1, m, 32); oi = __shfl_xor(i1, m, 32);
            if (ov < v1 || (ov == v1 && oi < i1)) { v1 = ov; i1 = oi; }
        }
        bv0[r] = v0; bi0[r] = i0; bv1[r] = v1; bi1[r] = i1;
    }

    // C/D layout: lanes 0-15 hold M = r, lanes 16-31 hold M = 8 + r
    if (nl == 0) {
        #pragma unroll
        for (int r = 0; r < 8; ++r) {
            int m = half * 8 + r;
            sBv[w * 32 + m]      = bv0[r]; sBi[w * 32 + m]      = bi0[r];
            sBv[w * 32 + m + 16] = bv1[r]; sBi[w * 32 + m + 16] = bi1[r];
        }
    }
    __syncthreads();

    // ---- cross-wave merge (each wave covered a disjoint code range) ----
    if (tid < TOK_PER_BLK) {
        float bv = sBv[tid]; int bi = sBi[tid];
        #pragma unroll
        for (int q = 1; q < 8; ++q) {
            float ov = sBv[q * 32 + tid]; int oi = sBi[q * 32 + tid];
            if (ov < bv || (ov == bv && oi < bi)) { bv = ov; bi = oi; }
        }
        sFin[tid] = bi;
    }
    __syncthreads();

    // ---- gather winning codebook rows (L2-resident) to output ----
    const float4* cb4 = (const float4*)cb;
    float4* out4 = (float4*)out;
    for (int e = tid; e < TOK_PER_BLK * (LATENT / 4); e += 256) {
        int row = e >> 7;
        int col = e & 127;
        out4[(size_t)(tok0 + row) * (LATENT / 4) + col] =
            cb4[(size_t)sFin[row] * (LATENT / 4) + col];
    }
}

// ---------------------------------------------------------------------------
extern "C" void kernel_launch(void* const* d_in, const int* in_sizes, int n_in,
                              void* d_out, int out_size, void* d_ws, size_t ws_size,
                              hipStream_t stream) {
    (void)in_sizes; (void)n_in; (void)out_size; (void)ws_size;
    const float* x  = (const float*)d_in[0];   // [32768, 512] f32
    const float* cb = (const float*)d_in[1];   // [8192, 512]  f32
    float* out = (float*)d_out;                // [32768, 512] f32

    // workspace: bf16 hi (8MB) | bf16 lo (8MB) | csq (32KB)  ~= 16.8 MB
    unsigned short* chi = (unsigned short*)d_ws;
    unsigned short* clo = chi + (size_t)KCODES * LATENT;
    float* csq = (float*)(clo + (size_t)KCODES * LATENT);

    vq_split_codebook<<<KCODES, 256, 0, stream>>>(cb, chi, clo, csq);

    const size_t shmem =
        (size_t)2 * TOK_PER_BLK * APAD * sizeof(unsigned short)  // A hi/lo
        + 8 * 32 * sizeof(float)                                 // sBv
        + 8 * 32 * sizeof(int)                                   // sBi
        + 32 * sizeof(int);                                      // sFin
    vq_argmin_gather<<<NTOK / TOK_PER_BLK, 256, shmem, stream>>>(
        x, cb, chi, clo, csq, out);
}
